// MultiHeadSelfAttention_82927228551388
// MI455X (gfx1250) — matrix operand order repaired and checked by reading, not compiled, not run on hardware
//
#include <hip/hip_runtime.h>
#include <hip/hip_bf16.h>

// ---------------------------------------------------------------------------
// MHSA on gfx1250: 3 WMMA kernels (QKV GEMM -> flash attention -> proj GEMM)
// bf16 inputs / f32 accumulation via v_wmma_f32_16x16x32_bf16.
// Flash-attention Q/K tiles staged with global_load_async_to_lds_b128.
// ---------------------------------------------------------------------------

typedef __attribute__((ext_vector_type(16))) __bf16 v16bf;
typedef __attribute__((ext_vector_type(8)))  float  v8f;

#define DIMC   1024
#define HEADS  16
#define HEADD  64
#define SEQN   2048
#define BATCH  2
#define TOK    (BATCH * SEQN)      // 4096 rows

__device__ __forceinline__ unsigned short f2bf(float x) {
    unsigned int u = __builtin_bit_cast(unsigned int, x);
    unsigned int r = u + 0x7FFFu + ((u >> 16) & 1u);   // round-to-nearest-even
    return (unsigned short)(r >> 16);
}
// RNE-round both floats, then v_perm_b32 merges the two high halves:
// dst = { hi[31:24], hi[23:16], lo[31:24], lo[23:16] } -> packed bf16x2
__device__ __forceinline__ unsigned int pack2bf(float lo, float hi) {
    unsigned int ulo = __builtin_bit_cast(unsigned int, lo);
    unsigned int uhi = __builtin_bit_cast(unsigned int, hi);
    unsigned int rlo = ulo + 0x7FFFu + ((ulo >> 16) & 1u);
    unsigned int rhi = uhi + 0x7FFFu + ((uhi >> 16) & 1u);
    return __builtin_amdgcn_perm(rhi, rlo, 0x07060302u);
}

// async 16-byte global->LDS copy (CDNA5 TDM-adjacent path, ASYNCcnt-tracked)
#define ASYNC_CP16(ldsPtr, gPtr)                                               \
    asm volatile("global_load_async_to_lds_b128 %0, %1, off"                   \
                 :: "v"((unsigned int)(unsigned long long)(ldsPtr)),           \
                    "v"((unsigned long long)(gPtr))                            \
                 : "memory")
#define ASYNC_WAIT() asm volatile("s_wait_asynccnt 0x0" ::: "memory")

struct Frag {
    union { unsigned int u[8]; v16bf v; };
};

// A fragment (16x32 bf16): base = &row[ kk + 16*hi ]; 16 contiguous K values.
__device__ __forceinline__ void load_a_frag(Frag& f, const unsigned short* base) {
    const unsigned int* p0 = (const unsigned int*)(base);
    f.u[0] = p0[0]; f.u[1] = p0[1]; f.u[2] = p0[2]; f.u[3] = p0[3];
    f.u[4] = p0[4]; f.u[5] = p0[5]; f.u[6] = p0[6]; f.u[7] = p0[7];
}
// B fragment (32x16 bf16): base = &colRow[ 16*hi ]; 16 contiguous K values.
__device__ __forceinline__ void load_b_frag(Frag& f, const unsigned short* base) {
    const unsigned int* p = (const unsigned int*)(base);
    f.u[0] = p[0]; f.u[1] = p[1]; f.u[2] = p[2]; f.u[3] = p[3];
    f.u[4] = p[4]; f.u[5] = p[5]; f.u[6] = p[6]; f.u[7] = p[7];
}
__device__ __forceinline__ v8f wmma_bf16(const Frag& a, const Frag& b, v8f c) {
    return __builtin_amdgcn_wmma_f32_16x16x32_bf16(false, a.v, false, b.v,
                                                   (short)0, c, false, false);
}

// ---------------------------------------------------------------------------
// Kernel 1: qkv = x @ W_qkv + b_qkv, scattered to Q/K/V in [B,H,N,64] bf16.
// Block tile 128x128, 8 waves of 64x32. K tiles of 32.
// ---------------------------------------------------------------------------
__global__ __launch_bounds__(256)
void mhsa_qkv_gemm(const float* __restrict__ X, const float* __restrict__ Wqkv,
                   const float* __restrict__ bqkv,
                   unsigned short* __restrict__ qws,
                   unsigned short* __restrict__ kws,
                   unsigned short* __restrict__ vws) {
    __shared__ unsigned short As[128 * 40];   // [row][k], stride 40 halves
    __shared__ unsigned short Bs[128 * 40];   // transposed: [n][k]

    const int t    = threadIdx.x;
    const int lane = t & 31;
    const int w    = t >> 5;
    const int wm   = w >> 2;          // 0..1 -> 64-row slab
    const int wn   = w & 3;           // 0..3 -> 32-col slab
    const int hi   = lane >> 4;
    const int ln   = lane & 15;
    const int rowBase = blockIdx.y * 128;
    const int colBase = blockIdx.x * 128;

    v8f acc[4][2];
    #pragma unroll
    for (int mi = 0; mi < 4; ++mi)
        #pragma unroll
        for (int ni = 0; ni < 2; ++ni) acc[mi][ni] = (v8f){};

    for (int k0 = 0; k0 < DIMC; k0 += 32) {
        __syncthreads();
        // stage A (fp32 -> bf16): 128x32
        #pragma unroll
        for (int j = 0; j < 4; ++j) {
            int idx = j * 256 + t;
            int row = idx >> 3, f4 = idx & 7;
            const float4 xv = ((const float4*)(X + (size_t)(rowBase + row) * DIMC + k0))[f4];
            unsigned int* dst = (unsigned int*)&As[row * 40];
            dst[f4 * 2]     = pack2bf(xv.x, xv.y);
            dst[f4 * 2 + 1] = pack2bf(xv.z, xv.w);
        }
        // stage B transposed, two k-rows per thread -> one b32 LDS store
        #pragma unroll
        for (int j = 0; j < 8; ++j) {
            int idx = j * 256 + t;
            int kp = idx >> 7, n = idx & 127;          // kp: 0..15 -> k = 2*kp
            const float* wp = Wqkv + (size_t)(k0 + 2 * kp) * 3072 + colBase + n;
            ((unsigned int*)&Bs[n * 40])[kp] = pack2bf(wp[0], wp[3072]);
        }
        if (k0 + 32 < DIMC)
            __builtin_prefetch(X + (size_t)(rowBase + (t >> 1)) * DIMC + k0 + 32 + (t & 1) * 16, 0, 1);
        __syncthreads();

        Frag af[4], bf2[2];
        #pragma unroll
        for (int mi = 0; mi < 4; ++mi)
            load_a_frag(af[mi], &As[(64 * wm + 16 * mi + ln) * 40 + 16 * hi]);
        #pragma unroll
        for (int ni = 0; ni < 2; ++ni)
            load_b_frag(bf2[ni], &Bs[(32 * wn + 16 * ni + ln) * 40 + 16 * hi]);
        #pragma unroll
        for (int mi = 0; mi < 4; ++mi)
            #pragma unroll
            for (int ni = 0; ni < 2; ++ni)
                acc[mi][ni] = wmma_bf16(af[mi], bf2[ni], acc[mi][ni]);
    }

    // epilogue: +bias, scatter into per-head Q/K/V bf16
    #pragma unroll
    for (int mi = 0; mi < 4; ++mi) {
        #pragma unroll
        for (int ni = 0; ni < 2; ++ni) {
            const int gcol = colBase + 32 * wn + 16 * ni + ln;
            const float bias = bqkv[gcol];
            const int which = gcol >> 10;
            const int cc = gcol & 1023;
            const int h  = cc >> 6, dd = cc & 63;
            unsigned short* dstp = (which == 0) ? qws : ((which == 1) ? kws : vws);
            #pragma unroll
            for (int r = 0; r < 8; ++r) {
                const int grow = rowBase + 64 * wm + 16 * mi + r + 8 * hi;
                const int b = grow >> 11, n = grow & 2047;
                const float val = acc[mi][ni][r] + bias;
                dstp[((size_t)(b * HEADS + h) * SEQN + n) * HEADD + dd] = f2bf(val);
            }
        }
    }
}

// ---------------------------------------------------------------------------
// Kernel 2: flash attention per (bh, 128 q-rows). Each wave owns 16 q-rows.
// Q/K tiles staged asynchronously to LDS; online softmax in registers.
// ---------------------------------------------------------------------------
__global__ __launch_bounds__(256)
void mhsa_flash_attn(const unsigned short* __restrict__ qws,
                     const unsigned short* __restrict__ kws,
                     const unsigned short* __restrict__ vws,
                     unsigned short* __restrict__ obf) {
    __shared__ unsigned short Qs[128 * 80];   // [qrow][d]
    __shared__ unsigned short Ks[64 * 80];    // [kpos][d]  (== B^T layout for QK^T)
    __shared__ unsigned short VT[64 * 80];    // [d][kpos]  (B^T layout for P@V)
    __shared__ unsigned short Ps[8 * 16 * 80];// per-wave P [16 qrows][64 kpos]

    const int t    = threadIdx.x;
    const int lane = t & 31;
    const int w    = t >> 5;
    const int hi   = lane >> 4;
    const int ln   = lane & 15;
    const int qblk = blockIdx.x;         // 0..15
    const int bh   = blockIdx.y;         // 0..31

    const unsigned short* Qp = qws + (size_t)bh * SEQN * HEADD + (size_t)qblk * 128 * HEADD;
    const unsigned short* Kp = kws + (size_t)bh * SEQN * HEADD;
    const unsigned short* Vp = vws + (size_t)bh * SEQN * HEADD;

    // stage Q block (128x64 bf16) asynchronously: 1024 x 16B transfers
    #pragma unroll
    for (int j = 0; j < 4; ++j) {
        int idx = j * 256 + t;
        int row = idx >> 3, q = idx & 7;
        ASYNC_CP16(&Qs[row * 80 + q * 8], Qp + row * HEADD + q * 8);
    }

    float m_r[8], l_r[8];
    v8f Oacc[4];
    #pragma unroll
    for (int r = 0; r < 8; ++r) { m_r[r] = -3.0e38f; l_r[r] = 0.0f; }
    #pragma unroll
    for (int oi = 0; oi < 4; ++oi) Oacc[oi] = (v8f){};

    const int qrow = w * 16 + ln;   // wave-local A row

    for (int kt = 0; kt < SEQN / 64; ++kt) {
        __syncthreads();
        const unsigned short* Kg = Kp + (size_t)kt * 64 * HEADD;
        const unsigned int*   Vg = (const unsigned int*)(Vp + (size_t)kt * 64 * HEADD);
        // K tile 64x64: async 512 x 16B transfers
        #pragma unroll
        for (int j = 0; j < 2; ++j) {
            int idx = j * 256 + t;
            int row = idx >> 3, q = idx & 7;
            ASYNC_CP16(&Ks[row * 80 + q * 8], Kg + row * HEADD + q * 8);
        }
        // V tile transposed (manual: transpose can't use async path)
        #pragma unroll
        for (int j = 0; j < 8; ++j) {
            int idx = j * 256 + t;
            int kp = idx >> 5, cu = idx & 31;
            unsigned int vv = Vg[kp * 32 + cu];
            int d = cu * 2;
            VT[d * 80 + kp]       = (unsigned short)(vv & 0xFFFFu);
            VT[(d + 1) * 80 + kp] = (unsigned short)(vv >> 16);
        }
        ASYNC_WAIT();               // all async (incl. Q on first iter) done
        __syncthreads();

        // S = Q @ K^T  (16 rows x 64 kpos per wave)
        v8f S[4];
        #pragma unroll
        for (int ni = 0; ni < 4; ++ni) S[ni] = (v8f){};
        #pragma unroll
        for (int kk = 0; kk < 64; kk += 32) {
            Frag aq;
            load_a_frag(aq, &Qs[qrow * 80 + kk + 16 * hi]);
            #pragma unroll
            for (int ni = 0; ni < 4; ++ni) {
                Frag bk;
                load_b_frag(bk, &Ks[(ni * 16 + ln) * 80 + kk + 16 * hi]);
                S[ni] = wmma_bf16(aq, bk, S[ni]);
            }
        }
        #pragma unroll
        for (int ni = 0; ni < 4; ++ni)
            #pragma unroll
            for (int r = 0; r < 8; ++r) S[ni][r] *= 0.125f;   // 1/sqrt(64)

        // online softmax: row = r + 8*hi, columns spread over 16 lanes
        float alpha[8];
        #pragma unroll
        for (int r = 0; r < 8; ++r) {
            float mx = fmaxf(fmaxf(S[0][r], S[1][r]), fmaxf(S[2][r], S[3][r]));
            #pragma unroll
            for (int msk = 1; msk < 16; msk <<= 1)
                mx = fmaxf(mx, __shfl_xor(mx, msk, 32));
            float mnew = fmaxf(m_r[r], mx);
            alpha[r] = __expf(m_r[r] - mnew);
            m_r[r] = mnew;
            l_r[r] *= alpha[r];
        }
        #pragma unroll
        for (int oi = 0; oi < 4; ++oi)
            #pragma unroll
            for (int r = 0; r < 8; ++r) Oacc[oi][r] *= alpha[r];

        // P = exp(S - m) -> per-wave LDS (A layout source); accumulate row sums
        float ps[8];
        #pragma unroll
        for (int r = 0; r < 8; ++r) ps[r] = 0.0f;
        #pragma unroll
        for (int ni = 0; ni < 4; ++ni)
            #pragma unroll
            for (int r = 0; r < 8; ++r) {
                float p = __expf(S[ni][r] - m_r[r]);
                ps[r] += p;
                Ps[w * 1280 + (r + 8 * hi) * 80 + ni * 16 + ln] = f2bf(p);
            }
        #pragma unroll
        for (int r = 0; r < 8; ++r) {
            #pragma unroll
            for (int msk = 1; msk < 16; msk <<= 1)
                ps[r] += __shfl_xor(ps[r], msk, 32);
            l_r[r] += ps[r];
        }
        __syncthreads();

        // O += P @ V
        #pragma unroll
        for (int kk = 0; kk < 64; kk += 32) {
            Frag ap;
            load_a_frag(ap, &Ps[w * 1280 + ln * 80 + kk + 16 * hi]);
            #pragma unroll
            for (int oi = 0; oi < 4; ++oi) {
                Frag bv;
                load_b_frag(bv, &VT[(oi * 16 + ln) * 80 + kk + 16 * hi]);
                Oacc[oi] = wmma_bf16(ap, bv, Oacc[oi]);
            }
        }
    }

    // normalize and write O token-major bf16 [4096][1024]
    const int b = bh >> 4, h = bh & 15;
    #pragma unroll
    for (int r = 0; r < 8; ++r) {
        const float rinv = 1.0f / l_r[r];
        const int row = b * SEQN + qblk * 128 + w * 16 + r + 8 * hi;
        #pragma unroll
        for (int oi = 0; oi < 4; ++oi) {
            const int col = h * HEADD + oi * 16 + ln;
            obf[(size_t)row * DIMC + col] = f2bf(Oacc[oi][r] * rinv);
        }
    }
}

// ---------------------------------------------------------------------------
// Kernel 3: out = O @ W_proj + b_proj  (fp32 out). Same tiling as kernel 1.
// ---------------------------------------------------------------------------
__global__ __launch_bounds__(256)
void mhsa_proj_gemm(const unsigned short* __restrict__ Ob,
                    const float* __restrict__ Wp, const float* __restrict__ bp,
                    float* __restrict__ out) {
    __shared__ unsigned short As[128 * 40];
    __shared__ unsigned short Bs[128 * 40];

    const int t    = threadIdx.x;
    const int lane = t & 31;
    const int w    = t >> 5;
    const int wm   = w >> 2;
    const int wn   = w & 3;
    const int hi   = lane >> 4;
    const int ln   = lane & 15;
    const int rowBase = blockIdx.y * 128;
    const int colBase = blockIdx.x * 128;

    v8f acc[4][2];
    #pragma unroll
    for (int mi = 0; mi < 4; ++mi)
        #pragma unroll
        for (int ni = 0; ni < 2; ++ni) acc[mi][ni] = (v8f){};

    for (int k0 = 0; k0 < DIMC; k0 += 32) {
        __syncthreads();
        #pragma unroll
        for (int j = 0; j < 8; ++j) {   // A tile: bf16 128x32 direct copy
            int idx = j * 256 + t;
            int row = idx >> 4, cu = idx & 15;
            ((unsigned int*)&As[row * 40])[cu] =
                ((const unsigned int*)(Ob + (size_t)(rowBase + row) * DIMC + k0))[cu];
        }
        #pragma unroll
        for (int j = 0; j < 8; ++j) {   // B transposed, paired fp32 -> bf16x2
            int idx = j * 256 + t;
            int kp = idx >> 7, n = idx & 127;
            const float* wp = Wp + (size_t)(k0 + 2 * kp) * DIMC + colBase + n;
            ((unsigned int*)&Bs[n * 40])[kp] = pack2bf(wp[0], wp[DIMC]);
        }
        if (k0 + 32 < DIMC)
            __builtin_prefetch(Ob + (size_t)(rowBase + (t >> 1)) * DIMC + k0 + 32, 0, 1);
        __syncthreads();

        Frag af[4], bf2[2];
        #pragma unroll
        for (int mi = 0; mi < 4; ++mi)
            load_a_frag(af[mi], &As[(64 * wm + 16 * mi + ln) * 40 + 16 * hi]);
        #pragma unroll
        for (int ni = 0; ni < 2; ++ni)
            load_b_frag(bf2[ni], &Bs[(32 * wn + 16 * ni + ln) * 40 + 16 * hi]);
        #pragma unroll
        for (int mi = 0; mi < 4; ++mi)
            #pragma unroll
            for (int ni = 0; ni < 2; ++ni)
                acc[mi][ni] = wmma_bf16(af[mi], bf2[ni], acc[mi][ni]);
    }

    #pragma unroll
    for (int mi = 0; mi < 4; ++mi)
        #pragma unroll
        for (int ni = 0; ni < 2; ++ni) {
            const int gcol = colBase + 32 * wn + 16 * ni + ln;
            const float bias = bp[gcol];
            #pragma unroll
            for (int r = 0; r < 8; ++r) {
                const int grow = rowBase + 64 * wm + 16 * mi + r + 8 * hi;
                out[(size_t)grow * DIMC + gcol] = acc[mi][ni][r] + bias;
            }
        }
}

// ---------------------------------------------------------------------------
extern "C" void kernel_launch(void* const* d_in, const int* in_sizes, int n_in,
                              void* d_out, int out_size, void* d_ws, size_t ws_size,
                              hipStream_t stream) {
    (void)in_sizes; (void)n_in; (void)out_size; (void)ws_size;
    const float* x     = (const float*)d_in[0];
    const float* W_qkv = (const float*)d_in[1];
    const float* b_qkv = (const float*)d_in[2];
    const float* W_prj = (const float*)d_in[3];
    const float* b_prj = (const float*)d_in[4];
    float* out = (float*)d_out;

    // workspace layout (bf16 halfwords): Q | K | V ([B,H,N,64]) | O ([4096,1024])
    const size_t HEAD_ELEMS = (size_t)BATCH * HEADS * SEQN * HEADD;  // 4 Mi
    unsigned short* ws  = (unsigned short*)d_ws;
    unsigned short* qws = ws;
    unsigned short* kws = ws + HEAD_ELEMS;
    unsigned short* vws = ws + 2 * HEAD_ELEMS;
    unsigned short* obf = ws + 3 * HEAD_ELEMS;

    mhsa_qkv_gemm<<<dim3(3072 / 128, TOK / 128), 256, 0, stream>>>(
        x, W_qkv, b_qkv, qws, kws, vws);
    mhsa_flash_attn<<<dim3(SEQN / 128, BATCH * HEADS), 256, 0, stream>>>(
        qws, kws, vws, obf);
    mhsa_proj_gemm<<<dim3(DIMC / 128, TOK / 128), 256, 0, stream>>>(
        obf, W_prj, b_prj, out);
}
